// HilbertAttentionTritonSimple_42185168781602
// MI455X (gfx1250) — matrix operand order repaired and checked by reading, not compiled, not run on hardware
//
#include <hip/hip_runtime.h>
#include <hip/hip_bf16.h>
#include <stdint.h>

// ---------------------------------------------------------------------------
// CDNA5 (gfx1250) WMMA types
// ---------------------------------------------------------------------------
typedef __bf16 bf16_t;
typedef bf16_t v16bf __attribute__((ext_vector_type(16)));
typedef float  v8f   __attribute__((ext_vector_type(8)));

union Frag16 {
    v16bf v;
    uint4 q[2];
    unsigned short u[16];
};

__device__ __forceinline__ unsigned short f2bf(float f) {
    union { float f; unsigned u; } x; x.f = f;
    unsigned r = 0x7FFFu + ((x.u >> 16) & 1u);   // round-to-nearest-even
    return (unsigned short)((x.u + r) >> 16);
}

// serpentine "hilbert" permutation for M = 4096 (g = 64)
__device__ __forceinline__ int hperm(int i) {
    int r = i >> 6, c = i & 63;
    return (r << 6) | ((r & 1) ? (63 - c) : c);
}

// ---------------------------------------------------------------------------
// Kernel 1/3: C[M,N] = A[M,K] * B[N,K]^T   (bf16 WMMA, f32 accumulate)
//   A: fp32 or bf16 (template), B: fp32 weights [N,K], C: fp32 or bf16.
// Block tile 128x128, K-step 32, 8 waves (4x2), each wave 32x64 (2x4 tiles).
// Pipeline: global->regs prefetch of next K-tile overlapped with WMMA stage.
// ---------------------------------------------------------------------------
#define BK      32
#define LDT_PAD 40   // LDS row stride in halves (32 data + 8 pad), 80B

template <bool A_BF16, bool C_BF16>
__global__ __launch_bounds__(256)
void gemm_bf16_wmma(const void* __restrict__ Aptr,
                    const float* __restrict__ Bw,
                    void* __restrict__ Cptr,
                    int M_, int N_, int K_)
{
    __shared__ unsigned short At[128 * LDT_PAD];
    __shared__ unsigned short Bt[128 * LDT_PAD];

    const int tid  = threadIdx.x;
    const int lane = tid & 31;
    const int wave = tid >> 5;
    const int half = lane >> 4;     // 0: rows 0-7 / K-lo,  1: rows 8-15 / K-hi
    const int l16  = lane & 15;

    const int nblock = blockIdx.x * 128;
    const int mblock = blockIdx.y * 128;

    const int wm = (wave & 3) * 32;   // wave M offset in block
    const int wn = (wave >> 2) * 64;  // wave N offset in block

    // per-thread fill assignment: 4 passes, each pass 1 row-quad
    const int frow = tid >> 3;        // 0..31 (row within 32-row pass group)
    const int fcol = (tid & 7) * 4;   // 0,4,...,28

    const v8f vzero = {};
    v8f acc[2][4];
#pragma unroll
    for (int i = 0; i < 2; ++i)
#pragma unroll
        for (int j = 0; j < 4; ++j) acc[i][j] = vzero;

    unsigned short ra[4][4], rb[4][4];

    auto loadTiles = [&](int k0) {
#pragma unroll
        for (int pass = 0; pass < 4; ++pass) {
            const int row = pass * 32 + frow;
            if constexpr (A_BF16) {
                const ushort4 u = *(const ushort4*)((const unsigned short*)Aptr +
                                      (size_t)(mblock + row) * K_ + k0 + fcol);
                ra[pass][0] = u.x; ra[pass][1] = u.y; ra[pass][2] = u.z; ra[pass][3] = u.w;
            } else {
                const float4 f = *(const float4*)((const float*)Aptr +
                                      (size_t)(mblock + row) * K_ + k0 + fcol);
                ra[pass][0] = f2bf(f.x); ra[pass][1] = f2bf(f.y);
                ra[pass][2] = f2bf(f.z); ra[pass][3] = f2bf(f.w);
            }
            const float4 g = *(const float4*)(Bw + (size_t)(nblock + row) * K_ + k0 + fcol);
            rb[pass][0] = f2bf(g.x); rb[pass][1] = f2bf(g.y);
            rb[pass][2] = f2bf(g.z); rb[pass][3] = f2bf(g.w);
        }
    };
    auto storeTiles = [&]() {
#pragma unroll
        for (int pass = 0; pass < 4; ++pass) {
            const int row = pass * 32 + frow;
            unsigned short* pa = &At[row * LDT_PAD + fcol];
            pa[0] = ra[pass][0]; pa[1] = ra[pass][1]; pa[2] = ra[pass][2]; pa[3] = ra[pass][3];
            unsigned short* pb = &Bt[row * LDT_PAD + fcol];
            pb[0] = rb[pass][0]; pb[1] = rb[pass][1]; pb[2] = rb[pass][2]; pb[3] = rb[pass][3];
        }
    };

    loadTiles(0);                         // prologue

    for (int k0 = 0; k0 < K_; k0 += BK) {
        storeTiles();
        __syncthreads();

        if (k0 + BK < K_) {
            loadTiles(k0 + BK);           // overlap next tile's global loads with WMMA
            if (k0 + 2 * BK < K_) {       // L2/L0 prefetch of the tile after next
                __builtin_prefetch((const char*)Bw +
                    ((size_t)(nblock + frow) * K_ + k0 + 2 * BK) * sizeof(float), 0, 3);
            }
        }

        // ---- fragments (CDNA5 16-bit A/B layouts, ISA §7.12.2)
        Frag16 afr[2], bfr[4];
#pragma unroll
        for (int mt = 0; mt < 2; ++mt) {
            const unsigned short* p = &At[(wm + mt * 16 + l16) * LDT_PAD];
            afr[mt].q[0] = *(const uint4*)(p + half * 8);        // K = half*8 + 0..7
            afr[mt].q[1] = *(const uint4*)(p + 16 + half * 8);   // K = 16 + half*8 + 0..7
        }
#pragma unroll
        for (int nt = 0; nt < 4; ++nt) {
            const unsigned short* p = &Bt[(wn + nt * 16 + l16) * LDT_PAD + half * 8];
            bfr[nt].q[0] = *(const uint4*)(p);                   // K = half*8 + 0..7
            bfr[nt].q[1] = *(const uint4*)(p + 16);              // K = 16 + half*8 + 0..7
        }
#pragma unroll
        for (int mt = 0; mt < 2; ++mt)
#pragma unroll
            for (int nt = 0; nt < 4; ++nt)
                acc[mt][nt] = __builtin_amdgcn_wmma_f32_16x16x32_bf16(
                    false, afr[mt].v, false, bfr[nt].v,
                    (short)0, acc[mt][nt], false, false);
        __syncthreads();
    }

    // ---- epilogue: lane col = l16, VGPR v -> row = half*8 + v
#pragma unroll
    for (int mt = 0; mt < 2; ++mt) {
#pragma unroll
        for (int nt = 0; nt < 4; ++nt) {
            const int ncol = nblock + wn + nt * 16 + l16;
#pragma unroll
            for (int v = 0; v < 8; ++v) {
                const int mrow = mblock + wm + mt * 16 + half * 8 + v;
                const float val = acc[mt][nt][v];
                if constexpr (C_BF16)
                    ((unsigned short*)Cptr)[(size_t)mrow * N_ + ncol] = f2bf(val);
                else
                    ((float*)Cptr)[(size_t)mrow * N_ + ncol] = val;
            }
        }
    }
}

// ---------------------------------------------------------------------------
// Kernel 2: segmented dilated attention in Hilbert order.
//   qkv: bf16 [B*M, 3*1024]  (cols: which*1024 + h*64 + d)
//   out: bf16 [B*M, 1024]    rows in Hilbert order (reference never un-permutes)
// One block per (b, h, seg). 8 waves; each wave owns 16 of the 128 queries.
// 64 dilated keys (DIL=2), hd=64.
// ---------------------------------------------------------------------------
#define KV_PAD 72   // row stride in halves (64 data + 8 pad), 144B

__global__ __launch_bounds__(256)
void hilbert_attn(const unsigned short* __restrict__ qkv,
                  unsigned short* __restrict__ out,
                  int Mtok)
{
    __shared__ unsigned short Kt[64 * KV_PAD];          // [key][d]
    __shared__ unsigned short Vt[64 * KV_PAD];          // [d][key]  (transposed V)
    __shared__ unsigned short Pt[128 * KV_PAD];         // per-wave attn [16 x 64]

    const int tid  = threadIdx.x;
    const int lane = tid & 31;
    const int wave = tid >> 5;
    const int half = lane >> 4;
    const int l16  = lane & 15;

    const int s = blockIdx.x & 31;
    const int h = (blockIdx.x >> 5) & 15;
    const int b = blockIdx.x >> 9;

    const size_t base = (size_t)b * Mtok * 3072;

    // ---- gather K (dilated) rows and transposed V rows into LDS
    {
        const int key = tid >> 2;             // 64 keys, 4 threads each
        const int c0  = (tid & 3) * 16;       // 16 halves per thread
        const int tok = hperm(s * 128 + key * 2);       // DIL = 2
        const unsigned short* krow = qkv + base + (size_t)tok * 3072 + 1024 + h * 64;
        const unsigned short* vrow = krow + 1024;

        *(uint4*)(&Kt[key * KV_PAD + c0])     = *(const uint4*)(krow + c0);
        *(uint4*)(&Kt[key * KV_PAD + c0 + 8]) = *(const uint4*)(krow + c0 + 8);

        union { uint4 q[2]; unsigned short u[16]; } vtmp;
        vtmp.q[0] = *(const uint4*)(vrow + c0);
        vtmp.q[1] = *(const uint4*)(vrow + c0 + 8);
#pragma unroll
        for (int e = 0; e < 16; ++e)
            Vt[(c0 + e) * KV_PAD + key] = vtmp.u[e];
    }
    __syncthreads();

    // ---- Q fragments straight from global (row = l16 of this wave's 16 queries)
    const int qtok = hperm(s * 128 + wave * 16 + l16);
    const unsigned short* qrow = qkv + base + (size_t)qtok * 3072 + h * 64;
    Frag16 qf[2];
#pragma unroll
    for (int ks = 0; ks < 2; ++ks) {
        qf[ks].q[0] = *(const uint4*)(qrow + ks * 32 + half * 8);
        qf[ks].q[1] = *(const uint4*)(qrow + ks * 32 + 16 + half * 8);
    }

    // ---- scores: 16q x 64k  (4 n-tiles x 2 k-steps)
    const v8f vzero = {};
    v8f sc[4];
#pragma unroll
    for (int nt = 0; nt < 4; ++nt) sc[nt] = vzero;
#pragma unroll
    for (int nt = 0; nt < 4; ++nt) {
#pragma unroll
        for (int ks = 0; ks < 2; ++ks) {
            Frag16 kf;
            const unsigned short* p = &Kt[(nt * 16 + l16) * KV_PAD + ks * 32 + half * 8];
            kf.q[0] = *(const uint4*)(p);
            kf.q[1] = *(const uint4*)(p + 16);
            sc[nt] = __builtin_amdgcn_wmma_f32_16x16x32_bf16(
                false, qf[ks].v, false, kf.v, (short)0, sc[nt], false, false);
        }
    }

    // ---- softmax across 64 keys: reduce over 4 tiles + 16 lanes of half-wave
    const float scale = 0.125f;   // hd^-0.5, hd = 64
#pragma unroll
    for (int v = 0; v < 8; ++v) {
        float m = -3.0e38f;
#pragma unroll
        for (int nt = 0; nt < 4; ++nt) m = fmaxf(m, sc[nt][v] * scale);
#pragma unroll
        for (int d = 1; d < 16; d <<= 1) m = fmaxf(m, __shfl_xor(m, d, 32));
        float p[4], ssum = 0.0f;
#pragma unroll
        for (int nt = 0; nt < 4; ++nt) { p[nt] = __expf(sc[nt][v] * scale - m); ssum += p[nt]; }
#pragma unroll
        for (int d = 1; d < 16; d <<= 1) ssum += __shfl_xor(ssum, d, 32);
        const float inv = 1.0f / ssum;
        const int row = wave * 16 + half * 8 + v;
#pragma unroll
        for (int nt = 0; nt < 4; ++nt)
            Pt[row * KV_PAD + nt * 16 + l16] = f2bf(p[nt] * inv);
    }
    // wave-private LDS region: within-wave DS ordering (DScnt) is sufficient.

    // ---- O = attn[16x64] * V[64x64]
    v8f oacc[4];
#pragma unroll
    for (int nt = 0; nt < 4; ++nt) oacc[nt] = vzero;
#pragma unroll
    for (int ks = 0; ks < 2; ++ks) {
        Frag16 pf;
        const unsigned short* pr = &Pt[(wave * 16 + l16) * KV_PAD + ks * 32];
        pf.q[0] = *(const uint4*)(pr + half * 8);
        pf.q[1] = *(const uint4*)(pr + 16 + half * 8);
#pragma unroll
        for (int nt = 0; nt < 4; ++nt) {
            Frag16 vf;
            const unsigned short* vr = &Vt[(nt * 16 + l16) * KV_PAD + ks * 32 + half * 8];
            vf.q[0] = *(const uint4*)(vr);
            vf.q[1] = *(const uint4*)(vr + 16);
            oacc[nt] = __builtin_amdgcn_wmma_f32_16x16x32_bf16(
                false, pf.v, false, vf.v, (short)0, oacc[nt], false, false);
        }
    }

    // ---- store O (bf16, rows in Hilbert order)
    const size_t obase = ((size_t)b * Mtok + s * 128 + wave * 16) * 1024 + h * 64;
#pragma unroll
    for (int nt = 0; nt < 4; ++nt) {
        const int dcol = nt * 16 + l16;
#pragma unroll
        for (int v = 0; v < 8; ++v) {
            const int row = half * 8 + v;
            out[obase + (size_t)row * 1024 + dcol] = f2bf(oacc[nt][v]);
        }
    }
}

// ---------------------------------------------------------------------------
extern "C" void kernel_launch(void* const* d_in, const int* in_sizes, int n_in,
                              void* d_out, int out_size, void* d_ws, size_t ws_size,
                              hipStream_t stream)
{
    (void)in_sizes; (void)n_in; (void)out_size; (void)ws_size;

    const float* x     = (const float*)d_in[0];   // [4, 4096, 1024]
    const float* w_qkv = (const float*)d_in[1];   // [3072, 1024]
    const float* w_out = (const float*)d_in[2];   // [1024, 1024]
    float* outp = (float*)d_out;                  // [4, 4096, 1024]

    const int B = 4, M = 4096, D = 1024;
    const int BM_ = B * M;                        // 16384

    unsigned short* qkv = (unsigned short*)d_ws;                  // bf16 [16384, 3072] (96 MB)
    unsigned short* ao  = qkv + (size_t)BM_ * 3 * D;              // bf16 [16384, 1024] (32 MB)

    // 1) qkv = x @ w_qkv^T  -> bf16 workspace       (A fp32, C bf16)
    dim3 g1(3 * D / 128, BM_ / 128);
    gemm_bf16_wmma<false, true><<<g1, 256, 0, stream>>>(x, w_qkv, qkv, BM_, 3 * D, D);

    // 2) segmented dilated attention in Hilbert order -> bf16 workspace
    hilbert_attn<<<dim3(B * 16 * 32), 256, 0, stream>>>(qkv, ao, M);

    // 3) out = attn_out @ w_out^T -> fp32 d_out     (A bf16, C fp32)
    dim3 g3(D / 128, BM_ / 128);
    gemm_bf16_wmma<true, false><<<g3, 256, 0, stream>>>(ao, w_out, outp, BM_, D, D);
}
